// AttentionBlock_47141561041264
// MI455X (gfx1250) — matrix operand order repaired and checked by reading, not compiled, not run on hardware
//
#include <hip/hip_runtime.h>
#include <hip/hip_bf16.h>
#include <stdint.h>

// ---------------- problem constants ----------------
constexpr int B_  = 8;
constexpr int C_  = 512;
constexpr int L_  = 1024;
constexpr int G_  = 32;
constexpr int CPG = 16;     // channels per group
constexpr int NH  = 8;
constexpr int CH  = 64;     // channels per head
constexpr int OC3 = 3 * C_; // 1536
#define EPS_GN 1e-5f
#define QK_SCALE 0.35355339059327373f   // 64^(-1/4)

typedef __attribute__((ext_vector_type(16))) __bf16 v16bf;
typedef __attribute__((ext_vector_type(8)))  float  v8f;
typedef __attribute__((ext_vector_type(8)))  int    v8i;
typedef __attribute__((ext_vector_type(2)))  int    v2i;

// ---- CDNA5 async global->LDS path (guarded; falls back to reg staging) ----
#if defined(__has_builtin)
#  if __has_builtin(__builtin_amdgcn_global_load_async_to_lds_b64) && \
      __has_builtin(__builtin_amdgcn_s_wait_asynccnt)
#    define HAVE_ASYNC_LDS 1
#  endif
#endif
#ifndef HAVE_ASYNC_LDS
#  define HAVE_ASYNC_LDS 0
#  warning "gfx1250 async global->LDS builtins not available; using reg-staged fallback"
#endif

#if HAVE_ASYNC_LDS
typedef __attribute__((address_space(1))) v2i gbl_v2i;
typedef __attribute__((address_space(3))) v2i lds_v2i;
static __device__ __forceinline__ void async_cp_b64(const void* g, void* l) {
    __builtin_amdgcn_global_load_async_to_lds_b64(
        (gbl_v2i*)g, (lds_v2i*)l, 0, 0);
}
#endif

static __device__ __forceinline__ uint16_t f2bf(float f) {
    uint32_t u = __float_as_uint(f);
    u += 0x7FFFu + ((u >> 16) & 1u);        // round-to-nearest-even
    return (uint16_t)(u >> 16);
}

// 16-bit A-matrix (16x32) K-pair base for vgpr r, lane-half lh (ISA 7.12.2)
static __device__ __forceinline__ int a_kbase(int r, int lh) {
    return (r < 4) ? (2 * r + 8 * lh) : (16 + 2 * (r - 4) + 8 * lh);
}

// ---------------- kernel 1: weight fp32 -> bf16 ----------------
__global__ void wcvt_kernel(const float* __restrict__ qkvw,
                            const float* __restrict__ projw,
                            uint16_t* __restrict__ oq,
                            uint16_t* __restrict__ op) {
    int i = blockIdx.x * blockDim.x + threadIdx.x;
    if (i < OC3 * C_) oq[i] = f2bf(qkvw[i]);
    if (i < C_ * C_)  op[i] = f2bf(projw[i]);
}

// ---------------- kernel 2: GroupNorm -> bf16 ----------------
__global__ void __launch_bounds__(256)
gn_kernel(const float* __restrict__ x, const float* __restrict__ gs,
          const float* __restrict__ gb, uint16_t* __restrict__ xn) {
    int b = blockIdx.x / G_, g = blockIdx.x % G_;
    const float* xp = x + ((size_t)b * C_ + (size_t)g * CPG) * L_;
    float s = 0.f, ss = 0.f;
    for (int e = threadIdx.x; e < CPG * L_; e += 256) {
        float v = xp[e]; s += v; ss += v * v;
    }
    __shared__ float rs[256], rq[256];
    rs[threadIdx.x] = s; rq[threadIdx.x] = ss;
    __syncthreads();
    for (int w = 128; w > 0; w >>= 1) {
        if (threadIdx.x < w) { rs[threadIdx.x] += rs[threadIdx.x + w];
                               rq[threadIdx.x] += rq[threadIdx.x + w]; }
        __syncthreads();
    }
    const float invN = 1.f / (float)(CPG * L_);
    float mu  = rs[0] * invN;
    float var = rq[0] * invN - mu * mu;
    float inv = rsqrtf(var + EPS_GN);
    uint16_t* xo = xn + ((size_t)b * C_ + (size_t)g * CPG) * L_;
    for (int e = threadIdx.x; e < CPG * L_; e += 256) {
        int c = g * CPG + (e >> 10);
        xo[e] = f2bf((xp[e] - mu) * inv * gs[c] + gb[c]);
    }
}

// ---------------- shared GEMM core (128x128x32 block, 8 waves) ----------------
// W: bf16 [M x 512] row-major, X: bf16 [512 x 1024] row-major (per batch).
// A tile double-buffered + async copied (if available); B tile reg-staged.
struct GemmAcc { v8f a[2][4]; };

static __device__ __forceinline__ void gemm_core(
        const uint16_t* __restrict__ Wg, const uint16_t* __restrict__ Xg,
        int mTile, int nTile, GemmAcc& A,
        uint16_t (*As)[128][36], uint16_t (*Bs)[36]) {
    const int tid = threadIdx.x;
    const int lane = tid & 31, w = tid >> 5;
    const int wm = w >> 1, wn = w & 1;
    const int lh = lane >> 4, ln = lane & 15;
    #pragma unroll
    for (int i = 0; i < 2; ++i)
        #pragma unroll
        for (int j = 0; j < 4; ++j) A.a[i][j] = (v8f){0.f,0.f,0.f,0.f,0.f,0.f,0.f,0.f};

    const int arow = tid >> 1, ac0 = (tid & 1) * 16;
    const uint16_t* agp = Wg + (size_t)(mTile + arow) * C_ + ac0;
    const int bcr = tid >> 3, bl0 = (tid & 7) * 16;
    const uint16_t* bgp = Xg + (size_t)bcr * L_ + nTile + bl0;

    // ---- prologue: bring in tile 0 ----
#if HAVE_ASYNC_LDS
    #pragma unroll
    for (int j = 0; j < 4; ++j)
        async_cp_b64(agp + 4 * j, &As[0][arow][ac0 + 4 * j]);
#else
    uint4 a0 = *(const uint4*)(agp);
    uint4 a1 = *(const uint4*)(agp + 8);
#endif
    uint4 b0 = *(const uint4*)(bgp);
    uint4 b1 = *(const uint4*)(bgp + 8);

    for (int kt = 0; kt < C_; kt += 32) {
        const int cur = (kt >> 5) & 1;
#if HAVE_ASYNC_LDS
        __builtin_amdgcn_s_wait_asynccnt(0);   // my A-tile(kt) landed in LDS
#endif
        __syncthreads();                       // everyone's landed; prev reads done
        { // store B regs (tile kt), transposed Bs[l][c]
            uint32_t dd[8] = {b0.x,b0.y,b0.z,b0.w,b1.x,b1.y,b1.z,b1.w};
            #pragma unroll
            for (int j = 0; j < 16; ++j)
                Bs[bl0 + j][bcr] = (uint16_t)(dd[j >> 1] >> ((j & 1) * 16));
        }
#if !HAVE_ASYNC_LDS
        { // store A regs (tile kt)
            uint32_t* dst = (uint32_t*)&As[cur][arow][ac0];
            dst[0] = a0.x; dst[1] = a0.y; dst[2] = a0.z; dst[3] = a0.w;
            dst[4] = a1.x; dst[5] = a1.y; dst[6] = a1.z; dst[7] = a1.w;
        }
#endif
        if (kt + 32 < C_) {                    // prefetch tile kt+1 (overlaps compute)
#if HAVE_ASYNC_LDS
            #pragma unroll
            for (int j = 0; j < 4; ++j)
                async_cp_b64(agp + kt + 32 + 4 * j, &As[cur ^ 1][arow][ac0 + 4 * j]);
#else
            a0 = *(const uint4*)(agp + kt + 32);
            a1 = *(const uint4*)(agp + kt + 32 + 8);
#endif
            b0 = *(const uint4*)(bgp + (size_t)(kt + 32) * L_);
            b1 = *(const uint4*)(bgp + (size_t)(kt + 32) * L_ + 8);
        }
        __syncthreads();                       // B(kt) visible to all

        v16bf afr[2], bfr[4];
        #pragma unroll
        for (int i = 0; i < 2; ++i) {
            int row = wm * 32 + i * 16 + ln;
            v8i t;
            #pragma unroll
            for (int r = 0; r < 8; ++r)
                t[r] = *(const uint32_t*)&As[cur][row][a_kbase(r, lh)];
            afr[i] = __builtin_bit_cast(v16bf, t);
        }
        #pragma unroll
        for (int j = 0; j < 4; ++j) {
            int col = wn * 64 + j * 16 + ln;
            v8i t;
            #pragma unroll
            for (int r = 0; r < 8; ++r)
                t[r] = *(const uint32_t*)&Bs[col][a_kbase(r, lh)];
            bfr[j] = __builtin_bit_cast(v16bf, t);
        }
        #pragma unroll
        for (int i = 0; i < 2; ++i)
            #pragma unroll
            for (int j = 0; j < 4; ++j)
                A.a[i][j] = __builtin_amdgcn_wmma_f32_16x16x32_bf16(
                    false, afr[i], false, bfr[j], (short)0, A.a[i][j], false, false);
    }
}

// ---------------- kernel 3: QKV GEMM + bias + q/k scale -> bf16 ----------------
__global__ void __launch_bounds__(256)
qkv_gemm_kernel(const uint16_t* __restrict__ Wq, const uint16_t* __restrict__ Xn,
                const float* __restrict__ bias, uint16_t* __restrict__ Qout) {
    __shared__ uint16_t As[2][128][36];
    __shared__ uint16_t Bs[128][36];
    int bx = blockIdx.x;
    int b = bx / ((OC3 / 128) * (L_ / 128));
    int rem = bx % ((OC3 / 128) * (L_ / 128));
    int mTile = (rem >> 3) * 128, nTile = (rem & 7) * 128;
    const uint16_t* Xg = Xn + (size_t)b * C_ * L_;
    uint16_t* Og = Qout + (size_t)b * OC3 * L_;

    GemmAcc A;
    gemm_core(Wq, Xg, mTile, nTile, A, As, Bs);

    const int lane = threadIdx.x & 31, w = threadIdx.x >> 5;
    const int wm = w >> 1, wn = w & 1, lh = lane >> 4, ln = lane & 15;
    #pragma unroll
    for (int i = 0; i < 2; ++i)
        #pragma unroll
        for (int j = 0; j < 4; ++j)
            #pragma unroll
            for (int r = 0; r < 8; ++r) {
                int o = mTile + wm * 32 + i * 16 + r + 8 * lh;
                int l = nTile + wn * 64 + j * 16 + ln;
                float v = A.a[i][j][r] + bias[o];
                if (o < 2 * C_) v *= QK_SCALE;   // scale q and k rows
                Og[(size_t)o * L_ + l] = f2bf(v);
            }
}

// ---------------- kernel 4: fused flash attention ----------------
__global__ void __launch_bounds__(256)
attn_kernel(const uint16_t* __restrict__ qkv, uint16_t* __restrict__ a_out) {
    __shared__ uint16_t Kt[128][66];   // [s][c] transposed K tile
    __shared__ uint16_t Vs[64][130];   // [c][s] natural V tile
    __shared__ uint16_t Ps[8][16][32]; // per-wave scratch (probs relayout / O transpose)

    int bh = blockIdx.x >> 3;             // b*NH + h
    int tBlk = (blockIdx.x & 7) * 128;
    int b = bh >> 3, h = bh & 7;
    const uint16_t* base = qkv + (size_t)b * OC3 * L_;
    const uint16_t* Qp = base + (size_t)(h * CH) * L_;
    const uint16_t* Kp = base + (size_t)(C_ + h * CH) * L_;
    const uint16_t* Vp = base + (size_t)(2 * C_ + h * CH) * L_;

    const int tid = threadIdx.x, lane = tid & 31, w = tid >> 5;
    const int lh = lane >> 4, ln = lane & 15;
    const int t0 = tBlk + w * 16;

    // Q fragments (A layout): row t = t0 + ln, K = c (gathered once)
    v16bf qf[2];
    #pragma unroll
    for (int f = 0; f < 2; ++f) {
        v8i tv;
        #pragma unroll
        for (int r = 0; r < 8; ++r) {
            int c0 = f * 32 + a_kbase(r, lh);
            uint32_t lo = Qp[(size_t)c0 * L_ + t0 + ln];
            uint32_t hi = Qp[(size_t)(c0 + 1) * L_ + t0 + ln];
            tv[r] = lo | (hi << 16);
        }
        qf[f] = __builtin_bit_cast(v16bf, tv);
    }

    float m_run[8], l_run[8];
    v8f oacc[4];
    #pragma unroll
    for (int r = 0; r < 8; ++r) { m_run[r] = -1e30f; l_run[r] = 0.f; }
    #pragma unroll
    for (int j = 0; j < 4; ++j) oacc[j] = (v8f){0.f,0.f,0.f,0.f,0.f,0.f,0.f,0.f};

    // ---- staging pipeline: preload s-block 0 into registers ----
    const int cc = tid >> 2, s0 = (tid & 3) * 32;
    const uint16_t* gk = Kp + (size_t)cc * L_ + s0;
    const uint16_t* gv = Vp + (size_t)cc * L_ + s0;
    uint4 kr[4], vr[4];
    #pragma unroll
    for (int q = 0; q < 4; ++q) {
        kr[q] = *(const uint4*)(gk + q * 8);
        vr[q] = *(const uint4*)(gv + q * 8);
    }

    for (int sb = 0; sb < L_; sb += 128) {
        __syncthreads();                 // previous block's LDS reads done
        #pragma unroll
        for (int q = 0; q < 4; ++q) {    // store K transposed, V natural
            uint32_t dd[4] = {kr[q].x, kr[q].y, kr[q].z, kr[q].w};
            #pragma unroll
            for (int j = 0; j < 8; ++j)
                Kt[s0 + q * 8 + j][cc] = (uint16_t)(dd[j >> 1] >> ((j & 1) * 16));
            uint32_t* vd = (uint32_t*)&Vs[cc][s0 + q * 8];
            vd[0] = vr[q].x; vd[1] = vr[q].y; vd[2] = vr[q].z; vd[3] = vr[q].w;
        }
        __syncthreads();
        if (sb + 128 < L_) {             // prefetch next s-block (overlaps compute)
            #pragma unroll
            for (int q = 0; q < 4; ++q) {
                kr[q] = *(const uint4*)(gk + sb + 128 + q * 8);
                vr[q] = *(const uint4*)(gv + sb + 128 + q * 8);
            }
        }

        for (int ss = 0; ss < 4; ++ss) {
            int so = ss * 32;
            // --- scores: two 16x16 tiles (s columns so..so+31), K-dim = c = 64 ---
            v8f sc[2];
            #pragma unroll
            for (int st = 0; st < 2; ++st) {
                int sofs = so + st * 16;
                v8i k0v, k1v;
                #pragma unroll
                for (int r = 0; r < 8; ++r) {
                    k0v[r] = *(const uint32_t*)&Kt[sofs + ln][a_kbase(r, lh)];
                    k1v[r] = *(const uint32_t*)&Kt[sofs + ln][32 + a_kbase(r, lh)];
                }
                v16bf kf0 = __builtin_bit_cast(v16bf, k0v);
                v16bf kf1 = __builtin_bit_cast(v16bf, k1v);
                v8f z = (v8f){0.f,0.f,0.f,0.f,0.f,0.f,0.f,0.f};
                v8f p = __builtin_amdgcn_wmma_f32_16x16x32_bf16(
                            false, qf[0], false, kf0, (short)0, z, false, false);
                sc[st] = __builtin_amdgcn_wmma_f32_16x16x32_bf16(
                            false, qf[1], false, kf1, (short)0, p, false, false);
            }
            // --- online softmax (fp32) ---
            float alpha[8];
            #pragma unroll
            for (int r = 0; r < 8; ++r) {
                float v = fmaxf(sc[0][r], sc[1][r]);
                #pragma unroll
                for (int msk = 1; msk < 16; msk <<= 1)
                    v = fmaxf(v, __shfl_xor(v, msk, 32));
                float mn = fmaxf(m_run[r], v);
                float al = __expf(m_run[r] - mn);
                m_run[r] = mn;
                float p0 = __expf(sc[0][r] - mn);
                float p1 = __expf(sc[1][r] - mn);
                sc[0][r] = p0; sc[1][r] = p1;
                float rsum = p0 + p1;
                #pragma unroll
                for (int msk = 1; msk < 16; msk <<= 1)
                    rsum += __shfl_xor(rsum, msk, 32);
                l_run[r] = l_run[r] * al + rsum;
                alpha[r] = al;
            }
            #pragma unroll
            for (int j = 0; j < 4; ++j)
                #pragma unroll
                for (int r = 0; r < 8; ++r) oacc[j][r] *= alpha[r];

            // --- relayout probs C->A through per-wave LDS scratch ---
            #pragma unroll
            for (int r = 0; r < 8; ++r) {
                Ps[w][r + 8 * lh][ln]      = f2bf(sc[0][r]);
                Ps[w][r + 8 * lh][ln + 16] = f2bf(sc[1][r]);
            }
            v8i pv;
            #pragma unroll
            for (int r = 0; r < 8; ++r)
                pv[r] = *(const uint32_t*)&Ps[w][ln][a_kbase(r, lh)];
            v16bf pf = __builtin_bit_cast(v16bf, pv);

            // --- O += P x V (4 c-tiles, K-dim = s = 32) ---
            #pragma unroll
            for (int j = 0; j < 4; ++j) {
                v8i vv;
                #pragma unroll
                for (int r = 0; r < 8; ++r)
                    vv[r] = *(const uint32_t*)&Vs[j * 16 + ln][so + a_kbase(r, lh)];
                v16bf vf = __builtin_bit_cast(v16bf, vv);
                oacc[j] = __builtin_amdgcn_wmma_f32_16x16x32_bf16(
                            false, pf, false, vf, (short)0, oacc[j], false, false);
            }
        }
    }
    // epilogue: normalize, transpose each O tile through LDS, store 16B chunks along t
    float inv[8];
    #pragma unroll
    for (int r = 0; r < 8; ++r) inv[r] = 1.f / l_run[r];
    uint16_t* ao = a_out + (size_t)b * C_ * L_;
    const int crow = lane & 15, tch = (lane >> 4) * 8;
    #pragma unroll
    for (int j = 0; j < 4; ++j) {
        #pragma unroll
        for (int r = 0; r < 8; ++r)
            Ps[w][r + 8 * lh][ln] = f2bf(oacc[j][r] * inv[r]);   // [t][c] bf16
        uint4 pack;
        uint16_t* tp = (uint16_t*)&pack;
        #pragma unroll
        for (int q = 0; q < 8; ++q) tp[q] = Ps[w][tch + q][crow]; // read column c
        int c = h * CH + j * 16 + crow;
        *(uint4*)&ao[(size_t)c * L_ + t0 + tch] = pack;           // 16B along t
    }
}

// ---------------- kernel 5: proj GEMM + bias + residual -> fp32 ----------------
__global__ void __launch_bounds__(256)
proj_gemm_kernel(const uint16_t* __restrict__ Wp, const uint16_t* __restrict__ Ain,
                 const float* __restrict__ bias, const float* __restrict__ x,
                 float* __restrict__ out) {
    __shared__ uint16_t As[2][128][36];
    __shared__ uint16_t Bs[128][36];
    int bx = blockIdx.x;
    int b = bx / ((C_ / 128) * (L_ / 128));
    int rem = bx % ((C_ / 128) * (L_ / 128));
    int mTile = (rem >> 3) * 128, nTile = (rem & 7) * 128;
    const uint16_t* Xg = Ain + (size_t)b * C_ * L_;

    GemmAcc A;
    gemm_core(Wp, Xg, mTile, nTile, A, As, Bs);

    const int lane = threadIdx.x & 31, w = threadIdx.x >> 5;
    const int wm = w >> 1, wn = w & 1, lh = lane >> 4, ln = lane & 15;
    #pragma unroll
    for (int i = 0; i < 2; ++i)
        #pragma unroll
        for (int j = 0; j < 4; ++j)
            #pragma unroll
            for (int r = 0; r < 8; ++r) {
                int o = mTile + wm * 32 + i * 16 + r + 8 * lh;
                int l = nTile + wn * 64 + j * 16 + ln;
                size_t idx = ((size_t)b * C_ + o) * L_ + l;
                out[idx] = x[idx] + A.a[i][j][r] + bias[o];
            }
}

// ---------------- launcher ----------------
extern "C" void kernel_launch(void* const* d_in, const int* in_sizes, int n_in,
                              void* d_out, int out_size, void* d_ws, size_t ws_size,
                              hipStream_t stream) {
    (void)in_sizes; (void)n_in; (void)out_size; (void)ws_size;
    const float* x     = (const float*)d_in[0];
    const float* gns   = (const float*)d_in[1];
    const float* gnb   = (const float*)d_in[2];
    const float* qkvw  = (const float*)d_in[3];
    const float* qkvb  = (const float*)d_in[4];
    const float* projw = (const float*)d_in[5];
    const float* projb = (const float*)d_in[6];
    float* out = (float*)d_out;

    uint16_t* ws      = (uint16_t*)d_ws;
    uint16_t* w_qkvw  = ws;                                   // 1536*512
    uint16_t* w_projw = w_qkvw + (size_t)OC3 * C_;            // 512*512
    uint16_t* w_qkv   = w_projw + (size_t)C_ * C_;            // 8*1536*1024
    uint16_t* w_xn    = w_qkv + (size_t)B_ * OC3 * L_;        // 8*512*1024 (xn, reused as 'a')

    hipLaunchKernelGGL(wcvt_kernel, dim3((OC3 * C_ + 255) / 256), dim3(256), 0, stream,
                       qkvw, projw, w_qkvw, w_projw);
    hipLaunchKernelGGL(gn_kernel, dim3(B_ * G_), dim3(256), 0, stream,
                       x, gns, gnb, w_xn);
    hipLaunchKernelGGL(qkv_gemm_kernel, dim3(B_ * (OC3 / 128) * (L_ / 128)), dim3(256), 0, stream,
                       w_qkvw, w_xn, qkvb, w_qkv);
    hipLaunchKernelGGL(attn_kernel, dim3(B_ * NH * (L_ / 128)), dim3(256), 0, stream,
                       w_qkv, w_xn /* reused as 'a' */);
    hipLaunchKernelGGL(proj_gemm_kernel, dim3(B_ * (C_ / 128) * (L_ / 128)), dim3(256), 0, stream,
                       w_projw, w_xn, projb, x, out);
}
